// TSDecoder_14396730376348
// MI455X (gfx1250) — matrix operand order repaired and checked by reading, not compiled, not run on hardware
//
#include <hip/hip_runtime.h>
#include <hip/hip_bf16.h>
#include <stdint.h>

// ---------------------------------------------------------------------------
// TSDecoder GRU recurrence for MI455X (gfx1250, wave32, WMMA 16x16x32 f16)
//
// B=256, FINALDIM=1024, INDIM=512, TS=128.
//   h0 = x @ W_lin.T + b_lin
//   repeat 128x: hn = GRUCell(xt, h); xt = h = hn  (xt==h for t>=1, x0==0)
// Weights are packed once into WMMA B-fragment layout (f16); hidden state
// ping-pongs in L2-resident workspace. Each step = one [256,512]x[512,3072]
// f16 WMMA GEMM + gate math. A-tiles staged via GLOBAL_LOAD_ASYNC_TO_LDS.
// ---------------------------------------------------------------------------

typedef _Float16 half16 __attribute__((ext_vector_type(16)));
typedef _Float16 half8  __attribute__((ext_vector_type(8)));
typedef float    f32x8  __attribute__((ext_vector_type(8)));

#define GRU_B   256
#define GRU_D   512
#define GRU_TS  128

__device__ __forceinline__ float sigmoid_fast(float x) {
    return 1.0f / (1.0f + __expf(-x));
}
__device__ __forceinline__ float tanh_fast(float x) {
    // 1 - 2/(e^{2x}+1): saturates correctly at +/-1 for large |x|
    return 1.0f - 2.0f / (__expf(2.0f * x) + 1.0f);
}

__device__ __forceinline__ half16 wmma_ldA(const _Float16* ap) {
    // A fragment 16x32 f16: lanes 0-15 hold K 0..15, lanes 16-31 hold
    // K 16..31; caller passes ap already offset by (lane<16?0:16).
    half8 lo = *(const half8*)ap;
    half8 hi = *(const half8*)(ap + 8);
    return __builtin_shufflevector(lo, hi, 0, 1, 2, 3, 4, 5, 6, 7,
                                   8, 9, 10, 11, 12, 13, 14, 15);
}

// Stage `n16` 16-byte chunks from global into LDS with the CDNA5 async path
// (GLOBAL_LOAD_ASYNC_TO_LDS_B128, tracked by ASYNCcnt). The low 32 bits of a
// flat __shared__ pointer are the LDS byte offset (aperture bits are [63:32]).
// Caller must __syncthreads() afterwards before other waves read the tile.
template <int N16>
__device__ __forceinline__ void async_stage_tile(const _Float16* __restrict__ gsrc,
                                                 _Float16* lds, int tid) {
    const uint32_t lbase = (uint32_t)(uintptr_t)lds;
    const uint64_t gbase = (uint64_t)(uintptr_t)gsrc;
#pragma unroll
    for (int i = 0; i < N16 / 128; ++i) {
        const int c = tid + i * 128;
        uint32_t loff = lbase + (uint32_t)c * 16u;
        uint64_t ga   = gbase + (uint64_t)c * 16u;
        asm volatile("global_load_async_to_lds_b128 %0, %1, off"
                     :: "v"(loff), "v"(ga) : "memory");
    }
    asm volatile("s_wait_asynccnt 0" ::: "memory");
}

// ---------------------------------------------------------------------------
// Weight packing: B fragment in the contiguous k order this code uses (same as A), per lane L:
//   N = L%16, K = h + (L<16 ? 0 : 16), h = 0..15 contiguous halfs.
// packed[((tile*KC + kc)*32 + lane)*16 + h] so each lane does one 32B load.
// ---------------------------------------------------------------------------

__global__ void pack_gate_weights(const float* __restrict__ W_ih,
                                  const float* __restrict__ W_hh,
                                  _Float16* __restrict__ packed) {
    // Combined N = 3072: [W_ih rows 0..1535 | W_hh rows 0..1535], K = 512.
    // tiles: 192 col-tiles x 16 k-chunks x 32 lanes
    int g = blockIdx.x * blockDim.x + threadIdx.x;
    if (g >= 192 * 16 * 32) return;
    int lane = g & 31;
    int kc   = (g >> 5) & 15;
    int ct   = g >> 9;
    int nc   = ct * 16 + (lane & 15);
    int k    = kc * 32 + ((lane & 16) ? 16 : 0);
    const float* src = (nc < 1536) ? (W_ih + (size_t)nc * 512 + k)
                                   : (W_hh + (size_t)(nc - 1536) * 512 + k);
    _Float16* dst = packed + ((size_t)g << 4);
#pragma unroll
    for (int h = 0; h < 16; ++h) dst[h] = (_Float16)src[h];
}

__global__ void pack_lin_weights(const float* __restrict__ W_lin,
                                 _Float16* __restrict__ packed) {
    // W_lin: [512,1024]; N = 512 (32 tiles), K = 1024 (32 chunks)
    int g = blockIdx.x * blockDim.x + threadIdx.x;
    if (g >= 32 * 32 * 32) return;
    int lane = g & 31;
    int kc   = (g >> 5) & 31;
    int ct   = g >> 10;
    int n    = ct * 16 + (lane & 15);
    int k    = kc * 32 + ((lane & 16) ? 16 : 0);
    const float* src = W_lin + (size_t)n * 1024 + k;
    _Float16* dst = packed + ((size_t)g << 4);
#pragma unroll
    for (int h = 0; h < 16; ++h) dst[h] = (_Float16)src[h];
}

__global__ void convert_x_f16(const float* __restrict__ x,
                              _Float16* __restrict__ x16) {
    int g = blockIdx.x * blockDim.x + threadIdx.x;
    if (g < GRU_B * 1024) x16[g] = (_Float16)x[g];
}

// ---------------------------------------------------------------------------
// h0 = x @ W_lin.T + b_lin   ([256,1024] x [1024->512])
// grid (16,8), block 128 (4 waves). Each wave: one 16x16 output tile.
// ---------------------------------------------------------------------------
__global__ __launch_bounds__(128) void h0_gemm(
    const _Float16* __restrict__ packedLin,
    const _Float16* __restrict__ x16,
    const float* __restrict__ b_lin,
    float* __restrict__ h32, _Float16* __restrict__ h16) {
    __shared__ alignas(16) _Float16 Atile[16 * 1024];  // 32 KB
    const int tid  = threadIdx.x;
    const int lane = tid & 31;
    const int wave = tid >> 5;
    const int rowbase = blockIdx.x * 16;
    const int ct = blockIdx.y * 4 + wave;  // 0..31

    // async-stage 16 contiguous rows of x16 into LDS (2048 x 16B)
    async_stage_tile<2048>(x16 + (size_t)rowbase * 1024, Atile, tid);
    __syncthreads();

    f32x8 acc = {};
    const int koff8 = (lane & 16) ? 16 : 0;
    const int r16   = lane & 15;
#pragma unroll 4
    for (int kc = 0; kc < 32; ++kc) {
        half16 a = wmma_ldA(&Atile[r16 * 1024 + kc * 32 + koff8]);
        half16 b = *(const half16*)(packedLin +
                    (((size_t)(ct * 32 + kc) * 32 + lane) << 4));
        acc = __builtin_amdgcn_wmma_f32_16x16x32_f16(
            false, a, false, b, (short)0, acc, false, false);
    }

    const int j  = ct * 16 + r16;
    const float bj = b_lin[j];
    const int m0 = rowbase + ((lane & 16) ? 8 : 0);
#pragma unroll
    for (int e = 0; e < 8; ++e) {
        float v = acc[e] + bj;
        h32[(size_t)(m0 + e) * GRU_D + j] = v;
        h16[(size_t)(m0 + e) * GRU_D + j] = (_Float16)v;
    }
}

// ---------------------------------------------------------------------------
// One GRU step: gi/gh GEMM (shared A = h) + gate math.
// grid (16,8), block 128 (4 waves); wave owns hidden col-tile ct (0..31).
// packedW col-tiles: gi_r=ct, gi_z=32+ct, gi_n=64+ct, gh_r=96+ct,
//                    gh_z=128+ct, gh_n=160+ct.
// FIRST=1 => x_t == 0: skip gi WMMAs (gi reduces to b_ih).
// ---------------------------------------------------------------------------
template <int FIRST>
__global__ __launch_bounds__(128) void gru_step(
    const _Float16* __restrict__ packedW,
    const _Float16* __restrict__ hsrc16,
    const float* __restrict__ hsrc32,
    float* __restrict__ hdst32, _Float16* __restrict__ hdst16,
    const float* __restrict__ b_ih, const float* __restrict__ b_hh,
    float* __restrict__ out, int t) {
    __shared__ alignas(16) _Float16 Atile[16 * GRU_D];  // 16 KB
    const int tid  = threadIdx.x;
    const int lane = tid & 31;
    const int wave = tid >> 5;
    const int rowbase = blockIdx.x * 16;
    const int ct = blockIdx.y * 4 + wave;  // 0..31

    // async-stage the h tile (16 contiguous rows, 1024 x 16B) into LDS
    async_stage_tile<1024>(hsrc16 + (size_t)rowbase * GRU_D, Atile, tid);
    __syncthreads();

    f32x8 a_ir = {}, a_iz = {}, a_in = {};
    f32x8 a_hr = {}, a_hz = {}, a_hn = {};
    const int koff8 = (lane & 16) ? 16 : 0;
    const int r16   = lane & 15;

#pragma unroll 4
    for (int kc = 0; kc < 16; ++kc) {
        half16 a = wmma_ldA(&Atile[r16 * GRU_D + kc * 32 + koff8]);

        // hint next K-chunk of the hot B stream into cache
        if (kc < 15)
            __builtin_prefetch(packedW + (((size_t)((96 + ct) * 16 + kc + 1) * 32
                                           + lane) << 4), 0, 1);

        const size_t base = ((size_t)kc * 32 + lane) << 4;
        half16 b0 = *(const half16*)(packedW + (((size_t)(96 + ct) * 16) << 9) + base);
        a_hr = __builtin_amdgcn_wmma_f32_16x16x32_f16(
            false, a, false, b0, (short)0, a_hr, false, false);
        half16 b1 = *(const half16*)(packedW + (((size_t)(128 + ct) * 16) << 9) + base);
        a_hz = __builtin_amdgcn_wmma_f32_16x16x32_f16(
            false, a, false, b1, (short)0, a_hz, false, false);
        half16 b2 = *(const half16*)(packedW + (((size_t)(160 + ct) * 16) << 9) + base);
        a_hn = __builtin_amdgcn_wmma_f32_16x16x32_f16(
            false, a, false, b2, (short)0, a_hn, false, false);

        if (!FIRST) {  // x_t == h for t>=1: same A, gi weight columns
            half16 b3 = *(const half16*)(packedW + (((size_t)ct * 16) << 9) + base);
            a_ir = __builtin_amdgcn_wmma_f32_16x16x32_f16(
                false, a, false, b3, (short)0, a_ir, false, false);
            half16 b4 = *(const half16*)(packedW + (((size_t)(32 + ct) * 16) << 9) + base);
            a_iz = __builtin_amdgcn_wmma_f32_16x16x32_f16(
                false, a, false, b4, (short)0, a_iz, false, false);
            half16 b5 = *(const half16*)(packedW + (((size_t)(64 + ct) * 16) << 9) + base);
            a_in = __builtin_amdgcn_wmma_f32_16x16x32_f16(
                false, a, false, b5, (short)0, a_in, false, false);
        }
    }

    // Gate math. D layout: element e -> M = e + (lane<16?0:8), N = lane%16.
    const int j = ct * 16 + r16;
    const float bir = b_ih[j], biz = b_ih[GRU_D + j], bin = b_ih[2 * GRU_D + j];
    const float bhr = b_hh[j], bhz = b_hh[GRU_D + j], bhn = b_hh[2 * GRU_D + j];
    const int m0 = rowbase + ((lane & 16) ? 8 : 0);
#pragma unroll
    for (int e = 0; e < 8; ++e) {
        const int m = m0 + e;
        float r = sigmoid_fast(a_ir[e] + bir + a_hr[e] + bhr);
        float z = sigmoid_fast(a_iz[e] + biz + a_hz[e] + bhz);
        float n = tanh_fast(a_in[e] + bin + r * (a_hn[e] + bhn));
        float hold = hsrc32[(size_t)m * GRU_D + j];
        float hn = (1.0f - z) * n + z * hold;
        hdst32[(size_t)m * GRU_D + j] = hn;
        hdst16[(size_t)m * GRU_D + j] = (_Float16)hn;
        out[(size_t)m * (GRU_TS * GRU_D) + (size_t)t * GRU_D + j] = hn;
    }
}

// ---------------------------------------------------------------------------
extern "C" void kernel_launch(void* const* d_in, const int* in_sizes, int n_in,
                              void* d_out, int out_size, void* d_ws, size_t ws_size,
                              hipStream_t stream) {
    (void)in_sizes; (void)n_in; (void)out_size; (void)ws_size;
    const float* x     = (const float*)d_in[0];  // [256,1024]
    const float* W_lin = (const float*)d_in[1];  // [512,1024]
    const float* b_lin = (const float*)d_in[2];  // [512]
    const float* W_ih  = (const float*)d_in[3];  // [1536,512]
    const float* W_hh  = (const float*)d_in[4];  // [1536,512]
    const float* b_ih  = (const float*)d_in[5];  // [1536]
    const float* b_hh  = (const float*)d_in[6];  // [1536]
    float* out = (float*)d_out;                  // [256, 128*512]

    char* ws = (char*)d_ws;
    size_t off = 0;
    _Float16* packedGate = (_Float16*)(ws + off); off += 192u * 16 * 32 * 16 * 2; // 3 MB
    _Float16* packedLin  = (_Float16*)(ws + off); off += 32u * 32 * 32 * 16 * 2;  // 1 MB
    _Float16* x16        = (_Float16*)(ws + off); off += (size_t)GRU_B * 1024 * 2;
    float*    h32[2];
    _Float16* h16[2];
    h32[0] = (float*)(ws + off);     off += (size_t)GRU_B * GRU_D * 4;
    h32[1] = (float*)(ws + off);     off += (size_t)GRU_B * GRU_D * 4;
    h16[0] = (_Float16*)(ws + off);  off += (size_t)GRU_B * GRU_D * 2;
    h16[1] = (_Float16*)(ws + off);

    // one-time preprocessing (cheap; weights land L2-resident in WMMA layout)
    pack_gate_weights<<<(192 * 16 * 32 + 255) / 256, 256, 0, stream>>>(
        W_ih, W_hh, packedGate);
    pack_lin_weights<<<(32 * 32 * 32 + 255) / 256, 256, 0, stream>>>(
        W_lin, packedLin);
    convert_x_f16<<<(GRU_B * 1024 + 255) / 256, 256, 0, stream>>>(x, x16);

    // h0 = x @ W_lin.T + b_lin
    h0_gemm<<<dim3(16, 8), 128, 0, stream>>>(packedLin, x16, b_lin,
                                             h32[0], h16[0]);

    // 128 serial GRU steps, ping-ponging hidden state buffers
    gru_step<1><<<dim3(16, 8), 128, 0, stream>>>(
        packedGate, h16[0], h32[0], h32[1], h16[1], b_ih, b_hh, out, 0);
    for (int t = 1; t < GRU_TS; ++t) {
        const int s = t & 1, d = s ^ 1;
        gru_step<0><<<dim3(16, 8), 128, 0, stream>>>(
            packedGate, h16[s], h32[s], h32[d], h16[d],
            b_ih, b_hh, out, t);
    }
}
